// activation_61143154426356
// MI455X (gfx1250) — matrix operand order repaired and checked by reading, not compiled, not run on hardware
//
#include <hip/hip_runtime.h>

#define B_DIM 32
#define S_DIM 2048
#define D_DIM 1024

typedef __attribute__((ext_vector_type(16))) __bf16 v16bf;
typedef __attribute__((ext_vector_type(8)))  float  v8f;

__device__ __forceinline__ unsigned short f2bf(float f) {
    unsigned int u = __float_as_uint(f);
    unsigned int lsb = (u >> 16) & 1u;
    return (unsigned short)((u + 0x7FFFu + lsb) >> 16);   // round-to-nearest-even
}

// wave-relative LDS byte offset of a __shared__ object (AS3 pointer value)
__device__ __forceinline__ unsigned lds_off_u32(const void* p) {
    return (unsigned)(unsigned long long)(__attribute__((address_space(3))) const char*)p;
}

// 128 bytes (one K-chunk row) memory -> LDS, 8x async b128, ASYNCcnt-tracked
__device__ __forceinline__ void async_fill_row(unsigned ldsoff, const unsigned short* g) {
    unsigned long long ga = (unsigned long long)g;
#pragma unroll
    for (int q = 0; q < 8; ++q) {
        asm volatile("global_load_async_to_lds_b128 %0, %1, off"
                     :: "v"(ldsoff + (unsigned)(q * 16)),
                        "v"(ga + (unsigned long long)(q * 16))
                     : "memory");
    }
}

// ---------------------------------------------------------------------------
// Kernel 1: per-(b,d) mean and t = sqrt(centered sum of squares); zero flags.
// ---------------------------------------------------------------------------
__global__ void corr_stats_kernel(const float* __restrict__ x,
                                  float* __restrict__ mean,
                                  float* __restrict__ tvec,
                                  int* __restrict__ kill) {
    __shared__ float redS[256];
    __shared__ float redQ[256];
    const int tid = threadIdx.x;
    const int dl  = tid & 63;
    const int sg  = tid >> 6;
    const int d0  = blockIdx.x * 64;
    const int b   = blockIdx.y;
    const float* xb = x + (size_t)b * S_DIM * D_DIM + d0 + dl;
    float s = 0.f, q = 0.f;
    for (int ss = sg; ss < S_DIM; ss += 4) {
        float v = xb[(size_t)ss * D_DIM];
        s += v; q += v * v;
    }
    redS[tid] = s; redQ[tid] = q;
    __syncthreads();
    if (tid < 64) {
        float st = redS[tid] + redS[tid + 64] + redS[tid + 128] + redS[tid + 192];
        float qt = redQ[tid] + redQ[tid + 64] + redQ[tid + 128] + redQ[tid + 192];
        float m   = st / (float)S_DIM;
        float css = qt - st * m;                 // sum((x-mean)^2), fp32 exact
        mean[b * D_DIM + d0 + tid] = m;
        tvec[b * D_DIM + d0 + tid] = sqrtf(fmaxf(css, 0.f));
        kill[b * D_DIM + d0 + tid] = 0;          // deterministic per-call init
    }
}

// ---------------------------------------------------------------------------
// Kernel 2: xt[b][d][s] = bf16(x[b][s][d] - mean[b][d])  (LDS 32x32 transpose)
// ---------------------------------------------------------------------------
__global__ void corr_center_transpose_kernel(const float* __restrict__ x,
                                             const float* __restrict__ mean,
                                             unsigned short* __restrict__ xt) {
    __shared__ float tile[32][33];
    const int tid = threadIdx.x;
    const int d0  = blockIdx.x * 32;
    const int s0  = blockIdx.y * 32;
    const int b   = blockIdx.z;
    {
        const int dl = tid & 31;
        const int sl = tid >> 5;                  // 0..7
        const float m = mean[b * D_DIM + d0 + dl];
        const float* xb = x + (size_t)b * S_DIM * D_DIM;
#pragma unroll
        for (int st = 0; st < 4; ++st) {
            int ss = s0 + sl + st * 8;
            tile[sl + st * 8][dl] = xb[(size_t)ss * D_DIM + d0 + dl] - m;
        }
    }
    __syncthreads();
    {
        const int sl = tid & 31;
        const int dl = tid >> 5;                  // 0..7
        unsigned short* xb = xt + (size_t)b * D_DIM * S_DIM;
#pragma unroll
        for (int st = 0; st < 4; ++st) {
            int dd = d0 + dl + st * 8;
            xb[(size_t)dd * S_DIM + s0 + sl] = f2bf(tile[sl][dl + st * 8]);
        }
    }
}

// ---------------------------------------------------------------------------
// Kernel 3: per-batch Xc^T * Xc via v_wmma_f32_16x16x32_bf16, threshold test,
// set kill flags. 128x128 tile / block; 8 waves each do 2x4 16x16 tiles.
// Double-buffered LDS filled with GLOBAL_LOAD_ASYNC_TO_LDS_B128 (ASYNCcnt):
// compute chunk c from buf[c&1] while chunk c+1 streams into buf[(c+1)&1].
// Upper-triangle blocks only (symmetry: a hit kills both columns i and j).
// ---------------------------------------------------------------------------
__global__ __launch_bounds__(256) void corr_gemm_mask_kernel(
        const unsigned short* __restrict__ xt,
        const float* __restrict__ tvec,
        int* __restrict__ kill) {
    const int tileJ = blockIdx.x;
    const int tileI = blockIdx.y;
    if (tileI > tileJ) return;                    // uniform per block: EXEC stays full
    const int b  = blockIdx.z;
    const int I0 = tileI * 128;
    const int J0 = tileJ * 128;

    // [buf][row][k]: rows 0-127 = A (I-block), rows 128-255 = B (J-block), 64 k each
    __shared__ unsigned short lds[2][256 * 64];   // 64 KB

    const int tid  = threadIdx.x;
    const int lane = tid & 31;
    const int wave = tid >> 5;
    const int wm = wave >> 1;                     // 0..3 -> M offset 32*wm
    const int wn = wave & 1;                      // 0..1 -> N offset 64*wn
    const int ln = lane & 15;
    const int hi = (lane >> 4) & 1;

    const unsigned short* xtb = xt + (size_t)b * D_DIM * S_DIM;

    v8f acc[2][4] = {};

    // one LDS row (128 bytes) per thread per chunk
    const int r = tid;
    const int rowg = (r < 128) ? (I0 + r) : (J0 + r - 128);
    const unsigned short* gsrc = xtb + (size_t)rowg * S_DIM;
    const unsigned ldsRow0 = lds_off_u32(&lds[0][r * 64]);
    const unsigned ldsRow1 = lds_off_u32(&lds[1][r * 64]);

    async_fill_row(ldsRow0, gsrc);                // prologue: chunk 0 -> buf 0

    const int nchunks = S_DIM / 64;
    for (int c = 0; c < nchunks; ++c) {
        // my rows of chunk c have landed; barrier -> ALL rows landed, and all
        // waves are done reading buf[(c+1)&1] (their chunk c-1 compute)
        asm volatile("s_wait_asynccnt 0x0" ::: "memory");
        __syncthreads();
        if (c + 1 < nchunks)
            async_fill_row((c & 1) ? ldsRow0 : ldsRow1,
                           gsrc + (size_t)(c + 1) * 64);

        const unsigned short* cur = &lds[c & 1][0];
        union Frag { uint4 u[2]; v16bf v; };
#pragma unroll
        for (int ks = 0; ks < 64; ks += 32) {
            Frag aF[2], bF[4];
#pragma unroll
            for (int ti = 0; ti < 2; ++ti) {
                // A 16x32 bf16: lane holds M=ln, K = hi*8 + (0..7) and +16..+23
                int mloc = wm * 32 + ti * 16 + ln;
                const unsigned short* p = &cur[mloc * 64 + ks + hi * 8];
                aF[ti].u[0] = *(const uint4*)(p);
                aF[ti].u[1] = *(const uint4*)(p + 16);
            }
#pragma unroll
            for (int tj = 0; tj < 4; ++tj) {
                // B 32x16 bf16: lane holds N=ln, K = hi*8 + (0..7) and +16..+23
                int nloc = wn * 64 + tj * 16 + ln;
                const unsigned short* p = &cur[(128 + nloc) * 64 + ks + hi * 8];
                bF[tj].u[0] = *(const uint4*)(p);
                bF[tj].u[1] = *(const uint4*)(p + 16);
            }
#pragma unroll
            for (int ti = 0; ti < 2; ++ti)
#pragma unroll
                for (int tj = 0; tj < 4; ++tj)
                    acc[ti][tj] = __builtin_amdgcn_wmma_f32_16x16x32_bf16(
                        false, aF[ti].v, false, bF[tj].v,
                        (short)0, acc[ti][tj], false, false);
        }
    }

    // threshold: |dot_ij| > 0.5 * t_i * t_j  (i != j) -> kill columns i and j
    const float* tb = tvec + b * D_DIM;
    int* kb = kill + b * D_DIM;
#pragma unroll
    for (int ti = 0; ti < 2; ++ti) {
#pragma unroll
        for (int tj = 0; tj < 4; ++tj) {
            int jg = J0 + wn * 64 + tj * 16 + ln;
            float tjv = tb[jg];
#pragma unroll
            for (int rr = 0; rr < 8; ++rr) {
                int ig = I0 + wm * 32 + ti * 16 + rr + hi * 8;   // C layout: M = r + 8*hi
                float av = fabsf(acc[ti][tj][rr]);
                if (ig != jg && av > 0.5f * tb[ig] * tjv) {
                    atomicOr(&kb[ig], 1);
                    atomicOr(&kb[jg], 1);
                }
            }
        }
    }
}

// ---------------------------------------------------------------------------
// Kernel 4: out = x * !kill[col]   (float4 streaming)
// ---------------------------------------------------------------------------
__global__ void corr_apply_kernel(const float* __restrict__ x,
                                  const int* __restrict__ kill,
                                  float* __restrict__ out) {
    const size_t idx = (size_t)blockIdx.x * blockDim.x + threadIdx.x;  // float4 idx
    float4 v = ((const float4*)x)[idx];
    const int b  = (int)(idx / ((size_t)S_DIM * (D_DIM / 4)));
    const int d4 = (int)(idx & (D_DIM / 4 - 1));
    const int4 k4 = ((const int4*)kill)[b * (D_DIM / 4) + d4];
    v.x = k4.x ? 0.f : v.x;
    v.y = k4.y ? 0.f : v.y;
    v.z = k4.z ? 0.f : v.z;
    v.w = k4.w ? 0.f : v.w;
    ((float4*)out)[idx] = v;
}

extern "C" void kernel_launch(void* const* d_in, const int* in_sizes, int n_in,
                              void* d_out, int out_size, void* d_ws, size_t ws_size,
                              hipStream_t stream) {
    (void)in_sizes; (void)n_in; (void)out_size; (void)ws_size;
    const float* x = (const float*)d_in[0];
    float* out = (float*)d_out;

    // workspace layout: mean | t | kill | xt(bf16, transposed)  ~= 128.4 MB
    float* mean = (float*)d_ws;
    float* tvec = mean + B_DIM * D_DIM;
    int*   kill = (int*)(tvec + B_DIM * D_DIM);
    unsigned short* xt = (unsigned short*)(kill + B_DIM * D_DIM);

    corr_stats_kernel<<<dim3(D_DIM / 64, B_DIM), 256, 0, stream>>>(x, mean, tvec, kill);
    corr_center_transpose_kernel<<<dim3(D_DIM / 32, S_DIM / 32, B_DIM), 256, 0, stream>>>(x, mean, xt);
    corr_gemm_mask_kernel<<<dim3(D_DIM / 128, D_DIM / 128, B_DIM), 256, 0, stream>>>(xt, tvec, kill);
    const size_t n4 = (size_t)B_DIM * S_DIM * D_DIM / 4;
    corr_apply_kernel<<<(unsigned)(n4 / 256), 256, 0, stream>>>(x, kill, out);
}
